// SelfAttention_30992484008565
// MI455X (gfx1250) — matrix operand order repaired and checked by reading, not compiled, not run on hardware
//
#include <hip/hip_runtime.h>

// ---------------------------------------------------------------------------
// Causal multi-head self-attention forward for MI455X (gfx1250, wave32, WMMA).
// Pipeline: cast(x,W->bf16, W transposed) -> QKV GEMM (wmma, TDM-staged,
//           double-buffered LDS) -> flash attention (wmma QK^T and PV,
//           online softmax, TDM-staged K tile) -> output GEMM (wmma, f32 out).
// Matrix math: v_wmma_f32_16x16x32_bf16 with fp32 accumulation.
// Data movement: tensor_load_to_lds (TDM, TENSORcnt) for GEMM/K-tile staging.
// ---------------------------------------------------------------------------

typedef __bf16 bf16;
typedef __attribute__((ext_vector_type(16))) __bf16 v16bf;
typedef __attribute__((ext_vector_type(8)))  __bf16 v8bf;
typedef __attribute__((ext_vector_type(8)))  float  v8f;
typedef __attribute__((ext_vector_type(4)))  unsigned int u32x4;
typedef __attribute__((ext_vector_type(8)))  int i32x8;
typedef __attribute__((ext_vector_type(4)))  int i32x4;

#define B_DIM   4
#define L_DIM   2048
#define D_DIM   1024
#define H_DIM   16
#define KD_DIM  64
#define M_TOT   (B_DIM * L_DIM)      // 8192
#define N_QKV   (3 * D_DIM)          // 3072

// ---------------- WMMA helpers -------------------------------------------

__device__ inline v8f wmma_bf16(v16bf a, v16bf b, v8f c) {
  // D = A(16x32) * B(32x16) + C, fp32 accum
  return __builtin_amdgcn_wmma_f32_16x16x32_bf16(
      /*neg_a=*/false, a, /*neg_b=*/false, b,
      /*c_mod=*/(short)0, c, /*reuse_a=*/false, /*reuse_b=*/false);
}

// A fragment (16x32 bf16), tile stored row-major [row][k] with leading dim ld.
// ISA layout: lanes 0-15 row=lane, elems 0..7 = K 0..7, elems 8..15 = K 16..23;
//             lanes 16-31 same rows, K ranges 8..15 / 24..31.
__device__ inline v16bf load_frag_a(const bf16* base, int ld) {
  const int lane = threadIdx.x & 31;
  const bf16* p = base + (lane & 15) * ld + ((lane >> 4) << 3);
  v8bf lo = *(const v8bf*)(p);        // K = koff + 0..7
  v8bf hi = *(const v8bf*)(p + 16);   // K = koff + 16..23
  v16bf a;
#pragma unroll
  for (int i = 0; i < 8; ++i) { a[i] = lo[i]; a[i + 8] = hi[i]; }
  return a;
}

// B fragment (32x16 bf16) from a tile stored as B^T: row-major [n][k], ld.
// ISA layout: lanes 0-15 col=lane, elems 0..7 = K 0..7, elems 8..15 = K 16..23;
//             lanes 16-31 same cols, K ranges 8..15 / 24..31.
__device__ inline v16bf load_frag_b(const bf16* base, int ld) {
  const int lane = threadIdx.x & 31;
  const bf16* p = base + (lane & 15) * ld + ((lane >> 4) << 3);
  v8bf lo = *(const v8bf*)(p);        // K = koff + 0..7
  v8bf hi = *(const v8bf*)(p + 16);   // K = koff + 16..23
  v16bf b;
#pragma unroll
  for (int i = 0; i < 8; ++i) { b[i] = lo[i]; b[i + 8] = hi[i]; }
  return b;
}

// ---------------- Tensor Data Mover staging -------------------------------
// 2-D tile DMA: tile_d1 rows of tile_d0 bf16 elements, global row stride
// stride0 (elements), written contiguously into LDS at lds_dst.
// D# layout per CDNA5 ISA (group0 128b, group1 256b).

#define RFL(x) __builtin_amdgcn_readfirstlane((int)(x))

__device__ inline void tdm_load_2d(bf16* lds_dst, const bf16* gsrc,
                                   unsigned tensor_d0, unsigned tensor_d1,
                                   unsigned stride0_elems,
                                   unsigned tile_d0, unsigned tile_d1) {
  const unsigned lds_addr = (unsigned)(unsigned long long)(size_t)lds_dst;
  const unsigned long long ga = (unsigned long long)(size_t)gsrc;

  u32x4 g0;
  g0[0] = (unsigned)RFL(1u);                          // count=1, user mode
  g0[1] = (unsigned)RFL(lds_addr);                    // lds_addr (bytes)
  g0[2] = (unsigned)RFL((unsigned)ga);                // global_addr[31:0]
  g0[3] = (unsigned)RFL(((unsigned)(ga >> 32) & 0x01FFFFFFu) | (2u << 30));
                                                      // global_addr[56:32]|type=2

  i32x8 g1;
  g1[0] = RFL(1u << 16);                              // wg_mask=0, data_size=2B
  g1[1] = RFL((tensor_d0 & 0xFFFFu) << 16);           // tensor_dim0[15:0]
  g1[2] = RFL((tensor_d0 >> 16) | ((tensor_d1 & 0xFFFFu) << 16));
  g1[3] = RFL((tensor_d1 >> 16) | (tile_d0 << 16));   // tile_dim0
  g1[4] = RFL(tile_d1);                               // tile_dim1 (tile_dim2=0)
  g1[5] = RFL(stride0_elems);                         // tensor_dim0_stride[31:0]
  g1[6] = 0;                                          // stride0[47:32], stride1 lo
  g1[7] = 0;                                          // stride1 hi

  i32x4 z4 = {0, 0, 0, 0};
#if defined(__clang_major__) && (__clang_major__ >= 23)
  i32x8 z8 = {0, 0, 0, 0, 0, 0, 0, 0};
  __builtin_amdgcn_tensor_load_to_lds(g0, g1, z4, z4, z8, 0);
#else
  __builtin_amdgcn_tensor_load_to_lds(g0, g1, z4, z4, 0);
#endif
}

// ---------------- Cast / transpose kernels --------------------------------

__global__ __launch_bounds__(256) void k_cast(const float* __restrict__ src,
                                              bf16* __restrict__ dst, int n) {
  int i = blockIdx.x * 256 + threadIdx.x;
  if (i < n) dst[i] = (bf16)src[i];
}

// dst[n*K + k] = (bf16) src[k*N + n]   (produces W^T in [N][K] layout)
__global__ __launch_bounds__(256) void k_transpose_cast(
    const float* __restrict__ src, bf16* __restrict__ dst, int K, int N) {
  int i = blockIdx.x * 256 + threadIdx.x;
  if (i >= K * N) return;
  int n = i / K;
  int k = i - n * K;
  dst[i] = (bf16)src[k * N + n];
}

// ---------------- Tiled bf16 GEMM:  C[M][N] = A[M][K] * Bt[N][K]^T --------
// 256 threads = 8 waves; block tile 128x128, BK = 32, double-buffered LDS.
// Wave 0 drives the TDM; s_wait_tensorcnt(2) overlaps next slab's DMA with
// the wait for the current one. Wave (wm,wn) in 4x2 grid computes a 32x64
// sub-tile = 2x4 wmma accumulators.

template <bool OUT_F32>
__global__ __launch_bounds__(256) void k_gemm(
    const bf16* __restrict__ A, const bf16* __restrict__ Bt,
    void* __restrict__ Cout, int M, int N, int K) {
  __shared__ bf16 smA[2][128 * 32];
  __shared__ bf16 smB[2][128 * 32];

  const int tid = threadIdx.x;
  const int lane = tid & 31, wid = tid >> 5;
  const int lane15 = lane & 15, hi = lane >> 4;
  const int waveM = wid >> 1, waveN = wid & 1;
  const int bm = blockIdx.y * 128, bn = blockIdx.x * 128;

  const bf16* Ablk = A + (size_t)bm * K;
  const bf16* Bblk = Bt + (size_t)bn * K;

  v8f acc[2][4];
#pragma unroll
  for (int m = 0; m < 2; ++m)
#pragma unroll
    for (int n = 0; n < 4; ++n)
#pragma unroll
      for (int r = 0; r < 8; ++r) acc[m][n][r] = 0.0f;

  // Prologue: DMA slab 0 into buffer 0 (2 TDM ops: A then B).
  if (wid == 0) {
    tdm_load_2d(smA[0], Ablk, (unsigned)K, (unsigned)(M - bm), (unsigned)K,
                32u, 128u);
    tdm_load_2d(smB[0], Bblk, (unsigned)K, (unsigned)(N - bn), (unsigned)K,
                32u, 128u);
  }

  for (int k0 = 0; k0 < K; k0 += 32) {
    const int cur = (k0 >> 5) & 1;
    __syncthreads();  // all waves done reading buffer (1-cur) from iter-1
    if (wid == 0) {
      if (k0 + 32 < K) {
        tdm_load_2d(smA[1 - cur], Ablk + (k0 + 32), (unsigned)K,
                    (unsigned)(M - bm), (unsigned)K, 32u, 128u);
        tdm_load_2d(smB[1 - cur], Bblk + (k0 + 32), (unsigned)K,
                    (unsigned)(N - bn), (unsigned)K, 32u, 128u);
        __builtin_amdgcn_s_wait_tensorcnt(2);  // buffer `cur` pair complete
      } else {
        __builtin_amdgcn_s_wait_tensorcnt(0);
      }
    }
    __syncthreads();  // buffer `cur` published to all waves

    v16bf a[2], b[4];
#pragma unroll
    for (int m = 0; m < 2; ++m)
      a[m] = load_frag_a(smA[cur] + (waveM * 32 + m * 16) * 32, 32);
#pragma unroll
    for (int n = 0; n < 4; ++n)
      b[n] = load_frag_b(smB[cur] + (waveN * 64 + n * 16) * 32, 32);
#pragma unroll
    for (int m = 0; m < 2; ++m)
#pragma unroll
      for (int n = 0; n < 4; ++n) acc[m][n] = wmma_bf16(a[m], b[n], acc[m][n]);
  }

  // Epilogue: C-fragment layout lane -> (row = r + hi*8, col = lane15)
#pragma unroll
  for (int m = 0; m < 2; ++m)
#pragma unroll
    for (int n = 0; n < 4; ++n)
#pragma unroll
      for (int r = 0; r < 8; ++r) {
        int row = bm + waveM * 32 + m * 16 + r + hi * 8;
        int col = bn + waveN * 64 + n * 16 + lane15;
        float v = acc[m][n][r];
        if (OUT_F32)
          ((float*)Cout)[(size_t)row * N + col] = v;
        else
          ((bf16*)Cout)[(size_t)row * N + col] = (bf16)v;
      }
}

// ---------------- Flash attention ----------------------------------------
// Grid: x = query tile (L/64 = 32), y = b*H + h (64). 128 threads = 4 waves.
// Each wave owns 16 query rows. Br = Bc = 64, Kd = 64.
// K tile staged via TDM (wave 0); V staged transposed by all threads.

#define VT_LD 72  // padded LDS stride for V^T tile (bank spread)

__global__ __launch_bounds__(128) void k_attention(
    const bf16* __restrict__ qkv, bf16* __restrict__ attn) {
  __shared__ bf16 smQ[64 * 64];
  __shared__ bf16 smK[64 * 64];
  __shared__ bf16 smVt[64 * VT_LD];
  __shared__ bf16 smP[64 * 64];

  const int tid = threadIdx.x;
  const int lane = tid & 31, w = tid >> 5;
  const int lane15 = lane & 15, hi = lane >> 4;
  const int ib = blockIdx.x;
  const int bh = blockIdx.y;
  const int b = bh >> 4, h = bh & 15;

  const size_t rowbase = (size_t)(b * L_DIM) * N_QKV;
  const int qcol = h * KD_DIM;
  const int kcol = D_DIM + h * KD_DIM;
  const int vcol = 2 * D_DIM + h * KD_DIM;

  // Stage Q tile [64 q][64 d]
  const bf16* qsrc = qkv + rowbase + (size_t)(ib * 64) * N_QKV + qcol;
#pragma unroll
  for (int i = tid; i < 512; i += 128) {
    int row = i >> 3, c8 = (i & 7) << 3;
    *(uint4*)(smQ + row * 64 + c8) =
        *(const uint4*)(qsrc + (size_t)row * N_QKV + c8);
  }

  float mrow[8], lrow[8];
  v8f O[4];
#pragma unroll
  for (int r = 0; r < 8; ++r) { mrow[r] = -3.0e38f; lrow[r] = 0.0f; }
#pragma unroll
  for (int n = 0; n < 4; ++n)
#pragma unroll
    for (int r = 0; r < 8; ++r) O[n][r] = 0.0f;

  for (int jb = 0; jb <= ib; ++jb) {
    __syncthreads();  // prior iteration done with smK/smVt; Q visible (jb==0)
    const bf16* ksrc = qkv + rowbase + (size_t)(jb * 64) * N_QKV + kcol;
    const bf16* vsrc = qkv + rowbase + (size_t)(jb * 64) * N_QKV + vcol;
    // K tile via Tensor Data Mover: 64 rows x 64 bf16, row stride 3072.
    if (w == 0) {
      tdm_load_2d(smK, ksrc, 64u, 64u, (unsigned)N_QKV, 64u, 64u);
    }
    // V staged transposed: smVt[dim][key]
#pragma unroll
    for (int i = tid; i < 4096; i += 128) {
      int key = i >> 6, dim = i & 63;
      smVt[dim * VT_LD + key] = vsrc[(size_t)key * N_QKV + dim];
    }
    if (w == 0) __builtin_amdgcn_s_wait_tensorcnt(0);
    __syncthreads();

    // S = Q * K^T  (per wave: 16 queries x 64 keys)
    v8f s[4];
#pragma unroll
    for (int n = 0; n < 4; ++n)
#pragma unroll
      for (int r = 0; r < 8; ++r) s[n][r] = 0.0f;
#pragma unroll
    for (int kk = 0; kk < 64; kk += 32) {
      v16bf aQ = load_frag_a(smQ + (w * 16) * 64 + kk, 64);
#pragma unroll
      for (int n = 0; n < 4; ++n) {
        v16bf bK = load_frag_b(smK + (n * 16) * 64 + kk, 64);
        s[n] = wmma_bf16(aQ, bK, s[n]);
      }
    }

    // Online softmax. Row = r + hi*8 spans lanes with same hi bit -> reduce
    // with xor masks 1,2,4,8 (wave32).
    const bool diag = (jb == ib);
#pragma unroll
    for (int r = 0; r < 8; ++r) {
      const int q = ib * 64 + w * 16 + r + hi * 8;
      float rm = -3.0e38f;
#pragma unroll
      for (int n = 0; n < 4; ++n) {
        float sv = s[n][r] * 0.125f;  // 1/sqrt(64)
        if (diag) {
          int kidx = jb * 64 + n * 16 + lane15;
          if (kidx > q) sv = -3.0e38f;
        }
        s[n][r] = sv;
        rm = fmaxf(rm, sv);
      }
#pragma unroll
      for (int msk = 1; msk < 16; msk <<= 1)
        rm = fmaxf(rm, __shfl_xor(rm, msk, 32));
      float mnew = fmaxf(mrow[r], rm);
      float corr = __expf(mrow[r] - mnew);
      mrow[r] = mnew;
      float rs = 0.0f;
#pragma unroll
      for (int n = 0; n < 4; ++n) {
        float p = __expf(s[n][r] - mnew);
        s[n][r] = p;
        rs += p;
      }
#pragma unroll
      for (int msk = 1; msk < 16; msk <<= 1) rs += __shfl_xor(rs, msk, 32);
      lrow[r] = lrow[r] * corr + rs;
#pragma unroll
      for (int n = 0; n < 4; ++n) O[n][r] *= corr;
    }

    // P: C-layout -> A-layout via wave-private LDS rows [w*16, w*16+16)
#pragma unroll
    for (int n = 0; n < 4; ++n)
#pragma unroll
      for (int r = 0; r < 8; ++r)
        smP[(w * 16 + r + hi * 8) * 64 + n * 16 + lane15] = (bf16)s[n][r];
    // (same-wave LDS RAW; DScnt wait inserted by compiler)

    // O += P * V   (A = P 16x64 keys, B = V^T tile [dim][key])
#pragma unroll
    for (int kk = 0; kk < 64; kk += 32) {
      v16bf aP = load_frag_a(smP + (w * 16) * 64 + kk, 64);
#pragma unroll
      for (int n = 0; n < 4; ++n) {
        v16bf bV = load_frag_b(smVt + (n * 16) * VT_LD + kk, VT_LD);
        O[n] = wmma_bf16(aP, bV, O[n]);
      }
    }
  }

  // Normalize and write attended [b, l, h*64 + d] as bf16
#pragma unroll
  for (int r = 0; r < 8; ++r) {
    float inv = 1.0f / lrow[r];
    int row = b * L_DIM + ib * 64 + w * 16 + r + hi * 8;
#pragma unroll
    for (int n = 0; n < 4; ++n) {
      attn[(size_t)row * D_DIM + h * KD_DIM + n * 16 + lane15] =
          (bf16)(O[n][r] * inv);
    }
  }
}

// ---------------- Host-side orchestration ---------------------------------

extern "C" void kernel_launch(void* const* d_in, const int* in_sizes, int n_in,
                              void* d_out, int out_size, void* d_ws,
                              size_t ws_size, hipStream_t stream) {
  const float* x    = (const float*)d_in[0];   // [4, 2048, 1024]
  const float* Wqkv = (const float*)d_in[1];   // [1024, 3072]
  const float* Wout = (const float*)d_in[2];   // [1024, 1024]
  float* out = (float*)d_out;                  // [4, 2048, 1024]

  // Workspace layout (bytes):
  //  xb     @ 0         : 8192*1024  bf16 = 16 MiB
  //  wqkvT  @ 16 MiB    : 3072*1024  bf16 =  6 MiB
  //  woutT  @ 22 MiB    : 1024*1024  bf16 =  2 MiB
  //  qkv    @ 24 MiB    : 8192*3072  bf16 = 48 MiB
  //  attn   @ 72 MiB    : 8192*1024  bf16 = 16 MiB   (total 88 MiB)
  char* ws = (char*)d_ws;
  bf16* xb    = (bf16*)(ws);
  bf16* wqkvT = (bf16*)(ws + (size_t)16 * 1024 * 1024);
  bf16* woutT = (bf16*)(ws + (size_t)22 * 1024 * 1024);
  bf16* qkv   = (bf16*)(ws + (size_t)24 * 1024 * 1024);
  bf16* attn  = (bf16*)(ws + (size_t)72 * 1024 * 1024);

  const int nx = M_TOT * D_DIM;  // 8388608
  k_cast<<<(nx + 255) / 256, 256, 0, stream>>>(x, xb, nx);
  k_transpose_cast<<<(D_DIM * N_QKV + 255) / 256, 256, 0, stream>>>(
      Wqkv, wqkvT, D_DIM, N_QKV);
  k_transpose_cast<<<(D_DIM * D_DIM + 255) / 256, 256, 0, stream>>>(
      Wout, woutT, D_DIM, D_DIM);

  // QKV projection: [8192,3072] = xb[8192,1024] * WqkvT^T
  k_gemm<false><<<dim3(N_QKV / 128, M_TOT / 128), 256, 0, stream>>>(
      xb, wqkvT, qkv, M_TOT, N_QKV, D_DIM);

  // Causal flash attention per (b,h, 64-query tile)
  k_attention<<<dim3(L_DIM / 64, B_DIM * H_DIM), 128, 0, stream>>>(qkv, attn);

  // Output projection: out[8192,1024] = attn * Wout  (fp32 out)
  k_gemm<true><<<dim3(D_DIM / 128, M_TOT / 128), 256, 0, stream>>>(
      attn, woutT, out, M_TOT, D_DIM, D_DIM);
}
